// SparseUnetV2_59219009077523
// MI455X (gfx1250) — matrix operand order repaired and checked by reading, compiled, not run on hardware
//
// ============================================================================
// Sparse 3D U-Net (SECOND-style) for MI455X / gfx1250 — compile-only target.
//
// Strategy (reasoned from MI455X specs, no runtime available):
//  * Conv layers = implicit GEMM on V_WMMA_F32_16X16X32_F16 (wave32, the
//    codegen-confirmed CDNA5 dense matrix path). M = 16 output voxels,
//    N = 16 output channels, K = 32 input channels per tap.
//  * Fragment loads are BRANCHLESS b128 vector loads: per lane the WMMA A/B
//    layouts are contiguous 16-byte chunks, and out-of-bounds / channel-pad
//    cases are handled by selecting the load *address* into a zeroed scratch
//    page (v_cndmask on pointer) instead of EXEC-masked scalar loads.
//  * Kernel is templated on Cin (16/32/64/128) so the K loop fully unrolls.
//  * Tap loop is fully de-div/mod-ed: nested td/th/tw loops with an
//    incrementally advanced weight pointer, and strides passed as log2
//    shifts so the transposed-conv divisibility test is a mask+shift
//    (round 2 showed s_mul_hi/s_cselect magic-division chains per tap).
//  * Activations kept as fp32 master + f16 shadow (f16 feeds WMMA; fp32 is
//    needed for residual adds, the channel-pair reduction in _ur, and the
//    final per-point gather). L1 f16 tensor = 27 MB -> fits the 192 MB L2,
//    so the 27 tap re-reads of each conv are L2 hits, not HBM traffic.
//  * Masked BN: atomic sum/sumsq/count reduction, then a fused
//    normalize*mask (+residual) (+ReLU) kernel that also emits the f16 copy.
//  * Transposed convs ("inv" blocks) are computed in gather form:
//    in = (out + pad_lo - tap)>>s with divisibility check, equivalent to
//    lax.conv_general_dilated(lhs_dilation=stride, weights flipped).
//  * Inputs are assumed flattened in setup_inputs() dict-insertion order:
//    [0]=voxel_features, [1]=coors, [2..89]=params leaves (w,g,b per block).
// ============================================================================
#include <hip/hip_runtime.h>
#include <hip/hip_bf16.h>

typedef __attribute__((ext_vector_type(16))) _Float16 v16h;
typedef __attribute__((ext_vector_type(8)))  float    v8f;
typedef __attribute__((ext_vector_type(4)))  unsigned u32x4;

#define BN_EPS 1e-3f

// ------------------------------- kernels -----------------------------------

union Frag16 { v16h v; u32x4 q[2]; };

// Implicit-GEMM masked 3D conv (forward or transposed-gather), f16 in, f32 out.
// One wave computes a 16(voxel) x 16(channel) output tile; 4 waves per block.
// CI = input channel count (compile-time; 16/32/64/128).
// Strides are passed as log2 shifts (all strides in this net are 1 or 2).
template<int CI>
__global__ __launch_bounds__(128) void conv3d_wmma(
    const _Float16* __restrict__ x, const _Float16* __restrict__ wT,
    const _Float16* __restrict__ zpage,          // >=32B of zeros
    float* __restrict__ yraw,
    int Di, int Hi, int Wi,
    int Do, int Ho, int Wo, int Co,
    int kd, int kh, int kw, int sdsh, int shsh, int swsh,
    int pd, int ph, int pw, int transposed)
{
  const int lane = threadIdx.x;          // 0..31
  const int half = lane >> 4;            // half-wave id
  const int col  = lane & 15;            // A row (M) for loads, D column (N) for store
  const int tileM = blockIdx.x * blockDim.y + threadIdx.y;
  const int n0 = blockIdx.y * 16;
  const int NvOut = Do * Ho * Wo;
  const int vbase = tileM * 16;
  if (vbase >= NvOut) return;            // uniform per wave

  const int vox = vbase + col;
  const bool rowOk = vox < NvOut;
  int od = 0, oh = 0, ow = 0;
  if (rowOk) { od = vox / (Ho * Wo); int r = vox - od * Ho * Wo; oh = r / Wo; ow = r - oh * Wo; }

  const int smd = (1 << sdsh) - 1, smh = (1 << shsh) - 1, smw = (1 << swsh) - 1;
  const long wstep = (long)Co * CI;
  const _Float16* wb = wT + ((long)n0 + col) * CI;  // tap 0; [tap][Cout][Cin] f16

  v8f acc = {0.f, 0.f, 0.f, 0.f, 0.f, 0.f, 0.f, 0.f};
  for (int td = 0; td < kd; ++td) {
    int id; bool okd = rowOk;
    if (!transposed) {
      id = (od << sdsh) - pd + td;
    } else {
      int jd = od + pd - td;
      okd = okd && jd >= 0 && (jd & smd) == 0;
      id = jd >> sdsh;
    }
    okd = okd && id >= 0 && id < Di;
    for (int th = 0; th < kh; ++th) {
      int ih; bool okh = okd;
      if (!transposed) {
        ih = (oh << shsh) - ph + th;
      } else {
        int jh = oh + ph - th;
        okh = okh && jh >= 0 && (jh & smh) == 0;
        ih = jh >> shsh;
      }
      okh = okh && ih >= 0 && ih < Hi;
      for (int tw = 0; tw < kw; ++tw, wb += wstep) {
        int iw; bool ok = okh;
        if (!transposed) {
          iw = (ow << swsh) - pw + tw;
        } else {
          int jw = ow + pw - tw;
          ok = ok && jw >= 0 && (jw & smw) == 0;
          iw = jw >> swsh;
        }
        ok = ok && iw >= 0 && iw < Wi;
        const _Float16* xb = x + ((long)(id * Hi + ih) * Wi + iw) * CI;

        #pragma unroll
        for (int c0 = 0; c0 < CI; c0 += 32) {
          Frag16 a, b;
          // A 16x32 f16 layout: lane M = lane&15; chunk0 K=[half*8,half*8+8),
          // chunk1 K=[16+half*8,+8). Both are 16B-aligned contiguous runs.
          const _Float16* pa0 = ok ? (xb + c0 + half * 8) : zpage;
          const _Float16* pa1 = (ok && (c0 + 32 <= CI)) ? (xb + c0 + 16 + half * 8) : zpage;
          // B 32x16 f16 layout: lane N = lane&15; chunk0 K=[half*8,half*8+8),
          // chunk1 K=[16+half*8,+8) (same as A).
          const _Float16* pb = ((c0 + half * 8) < CI) ? (wb + c0 + half * 8) : zpage;
          const _Float16* pb1 = (c0 + 32 <= CI) ? (wb + c0 + 16 + half * 8) : zpage;
          a.q[0] = *(const u32x4*)pa0;
          a.q[1] = *(const u32x4*)pa1;
          b.q[0] = *(const u32x4*)pb;
          b.q[1] = *(const u32x4*)pb1;
          acc = __builtin_amdgcn_wmma_f32_16x16x32_f16(false, a.v, false, b.v,
                                                       (short)0, acc, false, false);
        }
      }
    }
  }
  // C/D layout: vgpr r -> M = r + 8*half, N = lane&15
  #pragma unroll
  for (int r = 0; r < 8; ++r) {
    int v = vbase + r + 8 * half;
    if (v < NvOut) yraw[(long)v * Co + n0 + col] = acc[r];
  }
}

// Masked per-channel sum / sumsq / count reduction.
__global__ void bn_reduce(const float* __restrict__ y, const float* __restrict__ mask,
                          int Nv, int C, float* __restrict__ stats)
{
  __shared__ float sh1[256], sh2[256], shm[256];
  const int c = blockIdx.y;
  float s = 0.f, s2 = 0.f, sm = 0.f;
  for (int k = 0; k < 8; ++k) {
    int v = blockIdx.x * 2048 + k * 256 + threadIdx.x;
    if (v < Nv) {
      float m = mask[v];
      float val = y[(long)v * C + c] * m;
      s += val; s2 += val * val; sm += m;
    }
  }
  sh1[threadIdx.x] = s; sh2[threadIdx.x] = s2; shm[threadIdx.x] = sm;
  __syncthreads();
  for (int st = 128; st > 0; st >>= 1) {
    if (threadIdx.x < st) {
      sh1[threadIdx.x] += sh1[threadIdx.x + st];
      sh2[threadIdx.x] += sh2[threadIdx.x + st];
      shm[threadIdx.x] += shm[threadIdx.x + st];
    }
    __syncthreads();
  }
  if (threadIdx.x == 0) {
    atomicAdd(&stats[c], sh1[0]);
    atomicAdd(&stats[C + c], sh2[0]);
    if (blockIdx.y == 0) atomicAdd(&stats[2 * C], shm[0]);
  }
}

// Fused normalize*mask (+residual) (+relu); writes fp32 and/or f16 shadow.
__global__ void bn_apply(const float* __restrict__ yraw, const float* __restrict__ mask,
                         const float* __restrict__ stats, const float* __restrict__ g,
                         const float* __restrict__ b, const float* __restrict__ res,
                         int relu, float* __restrict__ out32, _Float16* __restrict__ out16,
                         int Nv, int C)
{
  long i = (long)blockIdx.x * blockDim.x + threadIdx.x;
  if (i >= (long)Nv * C) return;
  int c = (int)(i % C); long v = i / C;
  float n = stats[2 * C]; if (n < 1.f) n = 1.f;
  float mean = stats[c] / n;
  float var  = stats[C + c] / n - mean * mean;
  float val = ((yraw[i] - mean) * rsqrtf(var + BN_EPS) * g[c] + b[c]) * mask[v];
  if (res)  val += res[i];
  if (relu) val = fmaxf(val, 0.f);
  if (out32) out32[i] = val;
  if (out16) out16[i] = (_Float16)val;
}

// Convert fp32 weights (kd*kh*kw, Cin, Cout) -> f16 transposed (tap, Cout, CinPad).
__global__ void prep_w(const float* __restrict__ w, _Float16* __restrict__ wT,
                       int taps, int Ci, int CiPad, int Co)
{
  long i = (long)blockIdx.x * blockDim.x + threadIdx.x;
  long total = (long)taps * Co * CiPad;
  if (i >= total) return;
  int ci = (int)(i % CiPad); long r = i / CiPad;
  int co = (int)(r % Co); int t = (int)(r / Co);
  float v = (ci < Ci) ? w[((long)t * Ci + ci) * Co + co] : 0.f;
  wT[i] = (_Float16)v;
}

// Scatter voxel features into dense f16 grid (channel-padded) + mask.
__global__ void scatter_k(const float* __restrict__ vf, const int* __restrict__ coors,
                          int N, _Float16* __restrict__ x16, float* __restrict__ m1,
                          int H, int W, int Cp, int Cin)
{
  int i = blockIdx.x * blockDim.x + threadIdx.x;
  if (i >= N) return;
  int z = coors[i * 4 + 1], y = coors[i * 4 + 2], x = coors[i * 4 + 3];
  long idx = ((long)z * H + y) * W + x;
  m1[idx] = 1.f;
  for (int c = 0; c < Cin; ++c) x16[idx * Cp + c] = (_Float16)vf[i * Cin + c];
}

// m_out = (conv(mask, ones, stride, pad) > 0)
__global__ void mask_down_k(const float* __restrict__ mIn, float* __restrict__ mOut,
                            int Di, int Hi, int Wi, int Do, int Ho, int Wo,
                            int kd, int kh, int kw, int sd, int sh, int sw,
                            int pd, int ph, int pw)
{
  int v = blockIdx.x * blockDim.x + threadIdx.x;
  int Nv = Do * Ho * Wo;
  if (v >= Nv) return;
  int od = v / (Ho * Wo); int r = v - od * Ho * Wo; int oh = r / Wo; int ow = r - oh * Wo;
  float acc = 0.f;
  for (int td = 0; td < kd; ++td) { int id = od * sd - pd + td; if (id < 0 || id >= Di) continue;
    for (int th = 0; th < kh; ++th) { int ih = oh * sh - ph + th; if (ih < 0 || ih >= Hi) continue;
      for (int tw = 0; tw < kw; ++tw) { int iw = ow * sw - pw + tw; if (iw < 0 || iw >= Wi) continue;
        acc += mIn[((long)id * Hi + ih) * Wi + iw];
      } } }
  mOut[v] = (acc > 0.f) ? 1.f : 0.f;
}

// cat = concat([bot, trans], channel axis), f16.
__global__ void pack_cat_k(const _Float16* __restrict__ bot, const _Float16* __restrict__ trans,
                           _Float16* __restrict__ cat, int Nv, int C)
{
  long i = (long)blockIdx.x * blockDim.x + threadIdx.x;
  long tot = (long)Nv * 2 * C;
  if (i >= tot) return;
  int c = (int)(i % (2 * C)); long v = i / (2 * C);
  cat[i] = (c < C) ? bot[v * C + c] : trans[v * C + (c - C)];
}

// out = x_m + red, red[c] = cat[2c] + cat[2c+1] (cat = [bot, trans]).
__global__ void ur_finish_k(const float* __restrict__ xm, const float* __restrict__ bot,
                            const float* __restrict__ trans, float* __restrict__ out32,
                            _Float16* __restrict__ out16, int Nv, int C)
{
  long i = (long)blockIdx.x * blockDim.x + threadIdx.x;
  if (i >= (long)Nv * C) return;
  int c = (int)(i % C); long v = i / C;
  int j0 = 2 * c, j1 = 2 * c + 1;
  float r0 = (j0 < C) ? bot[v * C + j0] : trans[v * C + (j0 - C)];
  float r1 = (j1 < C) ? bot[v * C + j1] : trans[v * C + (j1 - C)];
  float val = xm[i] + r0 + r1;
  if (out32) out32[i] = val;
  if (out16) out16[i] = (_Float16)val;
}

// spatial = transpose NDHWC -> N(C*D)HW
__global__ void spatial_k(const float* __restrict__ src, float* __restrict__ dst,
                          int D, int H, int W, int C)
{
  int i = blockIdx.x * blockDim.x + threadIdx.x;
  int total = C * D * H * W;
  if (i >= total) return;
  int w = i % W; int r = i / W; int h = r % H; r /= H; int d = r % D; int c = r / D;
  dst[i] = src[(((long)d * H + h) * W + w) * C + c];
}

// Gather per-point features + cls/reg heads.
__global__ void heads_k(const float* __restrict__ xup1, const int* __restrict__ coors, int N,
                        const float* __restrict__ wc, const float* __restrict__ bc,
                        const float* __restrict__ wr, const float* __restrict__ br,
                        float* __restrict__ outCls, float* __restrict__ outReg,
                        float* __restrict__ outFeat, int H, int W)
{
  int i = blockIdx.x * blockDim.x + threadIdx.x;
  if (i >= N) return;
  int z = coors[i * 4 + 1], y = coors[i * 4 + 2], x = coors[i * 4 + 3];
  long idx = (((long)z * H + y) * W + x) * 16;
  float cls = bc[0];
  float r0 = br[0], r1 = br[1], r2 = br[2];
  for (int c = 0; c < 16; ++c) {
    float fv = xup1[idx + c];
    outFeat[(long)i * 16 + c] = fv;
    cls += fv * wc[c];
    r0 += fv * wr[c * 3 + 0]; r1 += fv * wr[c * 3 + 1]; r2 += fv * wr[c * 3 + 2];
  }
  outCls[i] = cls;
  outReg[i * 3 + 0] = r0; outReg[i * 3 + 1] = r1; outReg[i * 3 + 2] = r2;
}

// ------------------------------- host --------------------------------------

struct DimsT { int D, H, W; };
static inline int nvox(const DimsT& d) { return d.D * d.H * d.W; }

static inline void* bump(char* base, size_t& off, size_t bytes) {
  void* p = base + off;
  off += (bytes + 255) & ~(size_t)255;
  return p;
}

extern "C" void kernel_launch(void* const* d_in, const int* in_sizes, int n_in,
                              void* d_out, int out_size, void* d_ws, size_t ws_size,
                              hipStream_t stream)
{
  (void)out_size;
  if (n_in < 90 || !d_ws || !d_out) return;
  const float* vf = (const float*)d_in[0];
  const int* coors = (const int*)d_in[1];
  auto P = [&](int i) -> const float* { return (const float*)d_in[i]; };
  const int N = in_sizes[0] / 4;

  const DimsT L1{41, 152, 136}, L2{21, 76, 68}, L3{11, 38, 34}, L4{5, 19, 17}, LO{2, 19, 17};
  const int NV1 = nvox(L1), NV2 = nvox(L2), NV3 = nvox(L3), NV4 = nvox(L4), NVO = nvox(LO);

  // ---------------- workspace layout (bump allocation) ----------------
  char* base = (char*)d_ws; size_t off = 0;
  auto F32 = [&](long n) -> float* { return (float*)bump(base, off, (size_t)n * 4); };
  auto F16 = [&](long n) -> _Float16* { return (_Float16*)bump(base, off, (size_t)n * 2); };

  float* stats = F32(512);                 // sum[C], sumsq[C], count
  _Float16* zpage = F16(256);              // zeroed page for OOB fragment loads
  float* RAW = F32((long)NV1 * 16);        // shared raw conv output (max size)
  float* m1 = F32(NV1); float* m2 = F32(NV2); float* m3 = F32(NV3);
  float* m4 = F32(NV4); float* mo = F32(NVO);
  _Float16* xin16 = F16((long)NV1 * 16);
  _Float16* h16 = F16((long)NV1 * 16);
  float* x1_32 = F32((long)NV1 * 16); _Float16* x1_16 = F16((long)NV1 * 16);
  _Float16* h2_16 = F16((long)NV2 * 32); _Float16* t2_16 = F16((long)NV2 * 32);
  float* x2_32 = F32((long)NV2 * 32); _Float16* x2_16 = F16((long)NV2 * 32);
  _Float16* h3_16 = F16((long)NV3 * 64); _Float16* t3_16 = F16((long)NV3 * 64);
  float* x3_32 = F32((long)NV3 * 64); _Float16* x3_16 = F16((long)NV3 * 64);
  _Float16* h4_16 = F16((long)NV4 * 64); _Float16* t4_16 = F16((long)NV4 * 64);
  float* x4_32 = F32((long)NV4 * 64); _Float16* x4_16 = F16((long)NV4 * 64);
  float* out32 = F32((long)NVO * 128);
  // decoder L4
  _Float16* bb4_16 = F16((long)NV4 * 64);
  float* tr4_32 = F32((long)NV4 * 64); _Float16* tr4_16 = F16((long)NV4 * 64);
  _Float16* cat4_16 = F16((long)NV4 * 128);
  float* xm4_32 = F32((long)NV4 * 64);
  _Float16* ur4_16 = F16((long)NV4 * 64);
  float* xup4_32 = F32((long)NV3 * 64); _Float16* xup4_16 = F16((long)NV3 * 64);
  // decoder L3
  _Float16* bb3_16 = F16((long)NV3 * 64);
  float* tr3_32 = F32((long)NV3 * 64); _Float16* tr3_16 = F16((long)NV3 * 64);
  _Float16* cat3_16 = F16((long)NV3 * 128);
  float* xm3_32 = F32((long)NV3 * 64);
  _Float16* ur3_16 = F16((long)NV3 * 64);
  float* xup3_32 = F32((long)NV2 * 32); _Float16* xup3_16 = F16((long)NV2 * 32);
  // decoder L2
  _Float16* bb2_16 = F16((long)NV2 * 32);
  float* tr2_32 = F32((long)NV2 * 32); _Float16* tr2_16 = F16((long)NV2 * 32);
  _Float16* cat2_16 = F16((long)NV2 * 64);
  float* xm2_32 = F32((long)NV2 * 32);
  _Float16* ur2_16 = F16((long)NV2 * 32);
  float* xup2_32 = F32((long)NV1 * 16); _Float16* xup2_16 = F16((long)NV1 * 16);
  // decoder L1
  _Float16* bb1_16 = F16((long)NV1 * 16);
  float* tr1_32 = F32((long)NV1 * 16); _Float16* tr1_16 = F16((long)NV1 * 16);
  _Float16* cat1_16 = F16((long)NV1 * 32);
  float* xm1_32 = F32((long)NV1 * 16);
  _Float16* ur1_16 = F16((long)NV1 * 16);
  float* xup1_32 = F32((long)NV1 * 16);

  // weight arena: {param idx of w, taps, Cin, CinPad, Cout}
  struct WD { int idx, taps, Ci, CiPad, Co; };
  const WD wd[28] = {
    {2, 27, 4, 16, 16},  {5, 27, 16, 16, 16}, {8, 27, 16, 16, 32},
    {11, 27, 32, 32, 32},{14, 27, 32, 32, 32},{17, 27, 32, 32, 64},
    {20, 27, 64, 64, 64},{23, 27, 64, 64, 64},{26, 27, 64, 64, 64},
    {29, 27, 64, 64, 64},{32, 27, 64, 64, 64},{35, 3, 64, 64, 128},
    {38, 27, 64, 64, 64},{41, 27, 64, 64, 64},{44, 27, 128, 128, 64},
    {47, 27, 64, 64, 64},{50, 27, 64, 64, 64},{53, 27, 64, 64, 64},
    {56, 27, 128, 128, 64},{59, 27, 64, 64, 32},{62, 27, 32, 32, 32},
    {65, 27, 32, 32, 32},{68, 27, 64, 64, 32},{71, 27, 32, 32, 16},
    {74, 27, 16, 16, 16},{77, 27, 16, 16, 16},{80, 27, 32, 32, 16},
    {83, 27, 16, 16, 16}
  };
  _Float16* wts[28];
  for (int i = 0; i < 28; ++i)
    wts[i] = F16((long)wd[i].taps * wd[i].Co * wd[i].CiPad);

  if (off > ws_size) return;  // workspace too small: deterministic no-op

  // ---------------- launch helpers ----------------
  // stride args are log2 shifts (0 => stride 1, 1 => stride 2)
  auto conv = [&](const _Float16* x, const _Float16* wT, float* y,
                  DimsT in, int Ci, DimsT ou, int Co,
                  int kd, int kh, int kw, int ssd, int ssh, int ssw,
                  int pd, int ph, int pw, int tr) {
    int Nv = nvox(ou);
    dim3 grid((unsigned)((Nv + 63) / 64), (unsigned)(Co / 16));
    dim3 blk(32, 4);
    switch (Ci) {
      case 16:
        conv3d_wmma<16><<<grid, blk, 0, stream>>>(x, wT, zpage, y,
            in.D, in.H, in.W, ou.D, ou.H, ou.W, Co,
            kd, kh, kw, ssd, ssh, ssw, pd, ph, pw, tr); break;
      case 32:
        conv3d_wmma<32><<<grid, blk, 0, stream>>>(x, wT, zpage, y,
            in.D, in.H, in.W, ou.D, ou.H, ou.W, Co,
            kd, kh, kw, ssd, ssh, ssw, pd, ph, pw, tr); break;
      case 64:
        conv3d_wmma<64><<<grid, blk, 0, stream>>>(x, wT, zpage, y,
            in.D, in.H, in.W, ou.D, ou.H, ou.W, Co,
            kd, kh, kw, ssd, ssh, ssw, pd, ph, pw, tr); break;
      default:
        conv3d_wmma<128><<<grid, blk, 0, stream>>>(x, wT, zpage, y,
            in.D, in.H, in.W, ou.D, ou.H, ou.W, Co,
            kd, kh, kw, ssd, ssh, ssw, pd, ph, pw, tr); break;
    }
  };
  auto bn = [&](const float* y, const float* mask, int gi, const float* res, int relu,
                float* o32, _Float16* o16, int Nv, int C) {
    hipMemsetAsync(stats, 0, (size_t)(2 * C + 1) * 4, stream);
    bn_reduce<<<dim3((unsigned)((Nv + 2047) / 2048), (unsigned)C), dim3(256), 0, stream>>>(
        y, mask, Nv, C, stats);
    long tot = (long)Nv * C;
    bn_apply<<<dim3((unsigned)((tot + 255) / 256)), dim3(256), 0, stream>>>(
        y, mask, stats, P(gi), P(gi + 1), res, relu, o32, o16, Nv, C);
  };
  auto mdown = [&](const float* mi, float* mo_, DimsT in, DimsT ou,
                   int kd, int kh, int kw, int sd, int sh, int sw,
                   int pd, int ph, int pw) {
    int Nv = nvox(ou);
    mask_down_k<<<dim3((unsigned)((Nv + 255) / 256)), dim3(256), 0, stream>>>(
        mi, mo_, in.D, in.H, in.W, ou.D, ou.H, ou.W, kd, kh, kw, sd, sh, sw, pd, ph, pw);
  };
  auto pcat = [&](const _Float16* bot, const _Float16* tr, _Float16* cat, int Nv, int C) {
    long tot = (long)Nv * 2 * C;
    pack_cat_k<<<dim3((unsigned)((tot + 255) / 256)), dim3(256), 0, stream>>>(bot, tr, cat, Nv, C);
  };
  auto urf = [&](const float* xm, const float* bot, const float* tr,
                 float* o32, _Float16* o16, int Nv, int C) {
    long tot = (long)Nv * C;
    ur_finish_k<<<dim3((unsigned)((tot + 255) / 256)), dim3(256), 0, stream>>>(
        xm, bot, tr, o32, o16, Nv, C);
  };

  // ---------------- weight prep + zero page ----------------
  hipMemsetAsync(zpage, 0, 256 * sizeof(_Float16), stream);
  for (int i = 0; i < 28; ++i) {
    long tot = (long)wd[i].taps * wd[i].Co * wd[i].CiPad;
    prep_w<<<dim3((unsigned)((tot + 255) / 256)), dim3(256), 0, stream>>>(
        P(wd[i].idx), wts[i], wd[i].taps, wd[i].Ci, wd[i].CiPad, wd[i].Co);
  }

  // ---------------- scatter ----------------
  hipMemsetAsync(xin16, 0, (size_t)NV1 * 16 * sizeof(_Float16), stream);
  hipMemsetAsync(m1, 0, (size_t)NV1 * sizeof(float), stream);
  scatter_k<<<dim3((unsigned)((N + 255) / 256)), dim3(256), 0, stream>>>(
      vf, coors, N, xin16, m1, L1.H, L1.W, 16, 4);

  // ---------------- encoder ----------------
  conv(xin16, wts[0], RAW, L1, 16, L1, 16, 3,3,3, 0,0,0, 1,1,1, 0);   // conv_input
  bn(RAW, m1, 3, nullptr, 1, nullptr, h16, NV1, 16);
  conv(h16, wts[1], RAW, L1, 16, L1, 16, 3,3,3, 0,0,0, 1,1,1, 0);     // conv1
  bn(RAW, m1, 6, nullptr, 1, x1_32, x1_16, NV1, 16);

  mdown(m1, m2, L1, L2, 3,3,3, 2,2,2, 1,1,1);
  conv(x1_16, wts[2], RAW, L1, 16, L2, 32, 3,3,3, 1,1,1, 1,1,1, 0);   // conv2_0 (stride 2)
  bn(RAW, m2, 9, nullptr, 1, nullptr, h2_16, NV2, 32);
  conv(h2_16, wts[3], RAW, L2, 32, L2, 32, 3,3,3, 0,0,0, 1,1,1, 0);   // conv2_1
  bn(RAW, m2, 12, nullptr, 1, nullptr, t2_16, NV2, 32);
  conv(t2_16, wts[4], RAW, L2, 32, L2, 32, 3,3,3, 0,0,0, 1,1,1, 0);   // conv2_2
  bn(RAW, m2, 15, nullptr, 1, x2_32, x2_16, NV2, 32);

  mdown(m2, m3, L2, L3, 3,3,3, 2,2,2, 1,1,1);
  conv(x2_16, wts[5], RAW, L2, 32, L3, 64, 3,3,3, 1,1,1, 1,1,1, 0);   // conv3_0 (stride 2)
  bn(RAW, m3, 18, nullptr, 1, nullptr, h3_16, NV3, 64);
  conv(h3_16, wts[6], RAW, L3, 64, L3, 64, 3,3,3, 0,0,0, 1,1,1, 0);   // conv3_1
  bn(RAW, m3, 21, nullptr, 1, nullptr, t3_16, NV3, 64);
  conv(t3_16, wts[7], RAW, L3, 64, L3, 64, 3,3,3, 0,0,0, 1,1,1, 0);   // conv3_2
  bn(RAW, m3, 24, nullptr, 1, x3_32, x3_16, NV3, 64);

  mdown(m3, m4, L3, L4, 3,3,3, 2,2,2, 0,1,1);
  conv(x3_16, wts[8], RAW, L3, 64, L4, 64, 3,3,3, 1,1,1, 0,1,1, 0);   // conv4_0 (stride 2)
  bn(RAW, m4, 27, nullptr, 1, nullptr, h4_16, NV4, 64);
  conv(h4_16, wts[9], RAW, L4, 64, L4, 64, 3,3,3, 0,0,0, 1,1,1, 0);   // conv4_1
  bn(RAW, m4, 30, nullptr, 1, nullptr, t4_16, NV4, 64);
  conv(t4_16, wts[10], RAW, L4, 64, L4, 64, 3,3,3, 0,0,0, 1,1,1, 0);  // conv4_2
  bn(RAW, m4, 33, nullptr, 1, x4_32, x4_16, NV4, 64);

  mdown(m4, mo, L4, LO, 3,1,1, 2,1,1, 0,0,0);
  conv(x4_16, wts[11], RAW, L4, 64, LO, 128, 3,1,1, 1,0,0, 0,0,0, 0); // conv_out (stride 2,1,1)
  bn(RAW, mo, 36, nullptr, 1, out32, nullptr, NVO, 128);

  float* o_spatial = (float*)d_out;
  float* o_cls  = o_spatial + 128 * 2 * 19 * 17;
  float* o_reg  = o_cls + N;
  float* o_feat = o_reg + 3 * N;
  {
    int tot = 128 * 2 * 19 * 17;
    spatial_k<<<dim3((unsigned)((tot + 255) / 256)), dim3(256), 0, stream>>>(
        out32, o_spatial, 2, 19, 17, 128);
  }

  // ---------------- decoder ----------------
  // L4: _ur(x4, x4) -> inv4 -> xup4 (at L3)
  conv(x4_16, wts[12], RAW, L4, 64, L4, 64, 3,3,3, 0,0,0, 1,1,1, 0);
  bn(RAW, m4, 39, nullptr, 1, nullptr, bb4_16, NV4, 64);
  conv(bb4_16, wts[13], RAW, L4, 64, L4, 64, 3,3,3, 0,0,0, 1,1,1, 0);
  bn(RAW, m4, 42, x4_32, 1, tr4_32, tr4_16, NV4, 64);                 // residual + relu
  pcat(x4_16, tr4_16, cat4_16, NV4, 64);
  conv(cat4_16, wts[14], RAW, L4, 128, L4, 64, 3,3,3, 0,0,0, 1,1,1, 0);
  bn(RAW, m4, 45, nullptr, 1, xm4_32, nullptr, NV4, 64);
  urf(xm4_32, x4_32, tr4_32, nullptr, ur4_16, NV4, 64);
  conv(ur4_16, wts[15], RAW, L4, 64, L3, 64, 3,3,3, 1,1,1, 0,1,1, 1); // inv4 (transposed)
  bn(RAW, m3, 48, nullptr, 1, xup4_32, xup4_16, NV3, 64);

  // L3: _ur(x3, xup4) -> inv3 -> xup3 (at L2)
  conv(x3_16, wts[16], RAW, L3, 64, L3, 64, 3,3,3, 0,0,0, 1,1,1, 0);
  bn(RAW, m3, 51, nullptr, 1, nullptr, bb3_16, NV3, 64);
  conv(bb3_16, wts[17], RAW, L3, 64, L3, 64, 3,3,3, 0,0,0, 1,1,1, 0);
  bn(RAW, m3, 54, x3_32, 1, tr3_32, tr3_16, NV3, 64);
  pcat(xup4_16, tr3_16, cat3_16, NV3, 64);
  conv(cat3_16, wts[18], RAW, L3, 128, L3, 64, 3,3,3, 0,0,0, 1,1,1, 0);
  bn(RAW, m3, 57, nullptr, 1, xm3_32, nullptr, NV3, 64);
  urf(xm3_32, xup4_32, tr3_32, nullptr, ur3_16, NV3, 64);
  conv(ur3_16, wts[19], RAW, L3, 64, L2, 32, 3,3,3, 1,1,1, 1,1,1, 1); // inv3 (transposed)
  bn(RAW, m2, 60, nullptr, 1, xup3_32, xup3_16, NV2, 32);

  // L2: _ur(x2, xup3) -> inv2 -> xup2 (at L1)
  conv(x2_16, wts[20], RAW, L2, 32, L2, 32, 3,3,3, 0,0,0, 1,1,1, 0);
  bn(RAW, m2, 63, nullptr, 1, nullptr, bb2_16, NV2, 32);
  conv(bb2_16, wts[21], RAW, L2, 32, L2, 32, 3,3,3, 0,0,0, 1,1,1, 0);
  bn(RAW, m2, 66, x2_32, 1, tr2_32, tr2_16, NV2, 32);
  pcat(xup3_16, tr2_16, cat2_16, NV2, 32);
  conv(cat2_16, wts[22], RAW, L2, 64, L2, 32, 3,3,3, 0,0,0, 1,1,1, 0);
  bn(RAW, m2, 69, nullptr, 1, xm2_32, nullptr, NV2, 32);
  urf(xm2_32, xup3_32, tr2_32, nullptr, ur2_16, NV2, 32);
  conv(ur2_16, wts[23], RAW, L2, 32, L1, 16, 3,3,3, 1,1,1, 1,1,1, 1); // inv2 (transposed)
  bn(RAW, m1, 72, nullptr, 1, xup2_32, xup2_16, NV1, 16);

  // L1: _ur(x1, xup2) -> conv5 -> xup1
  conv(x1_16, wts[24], RAW, L1, 16, L1, 16, 3,3,3, 0,0,0, 1,1,1, 0);
  bn(RAW, m1, 75, nullptr, 1, nullptr, bb1_16, NV1, 16);
  conv(bb1_16, wts[25], RAW, L1, 16, L1, 16, 3,3,3, 0,0,0, 1,1,1, 0);
  bn(RAW, m1, 78, x1_32, 1, tr1_32, tr1_16, NV1, 16);
  pcat(xup2_16, tr1_16, cat1_16, NV1, 16);
  conv(cat1_16, wts[26], RAW, L1, 32, L1, 16, 3,3,3, 0,0,0, 1,1,1, 0);
  bn(RAW, m1, 81, nullptr, 1, xm1_32, nullptr, NV1, 16);
  urf(xm1_32, xup2_32, tr1_32, nullptr, ur1_16, NV1, 16);
  conv(ur1_16, wts[27], RAW, L1, 16, L1, 16, 3,3,3, 0,0,0, 1,1,1, 0); // conv5
  bn(RAW, m1, 84, nullptr, 1, xup1_32, nullptr, NV1, 16);

  // ---------------- heads ----------------
  heads_k<<<dim3((unsigned)((N + 255) / 256)), dim3(256), 0, stream>>>(
      xup1_32, coors, N, P(86), P(87), P(88), P(89), o_cls, o_reg, o_feat, L1.H, L1.W);
}
